// GraphAttentionNetwork_446676598799
// MI455X (gfx1250) — matrix operand order repaired and checked by reading, not compiled, not run on hardware
//
#include <hip/hip_runtime.h>

typedef __attribute__((ext_vector_type(16))) __bf16        v16bf;
typedef __attribute__((ext_vector_type(8)))  float         v8f;
typedef __attribute__((ext_vector_type(4)))  unsigned int  u32x4;
typedef __attribute__((ext_vector_type(4)))  float         f32x4;

__device__ __forceinline__ unsigned short f2bf(float f) {
    unsigned int u = __float_as_uint(f);
    u += 0x7FFFu + ((u >> 16) & 1u);   // round-to-nearest-even
    return (unsigned short)(u >> 16);
}

__device__ __forceinline__ void atomicMaxF(float* addr, float val) {
    if (val >= 0.0f) atomicMax((int*)addr, __float_as_int(val));
    else             atomicMin((unsigned int*)addr, __float_as_uint(val));
}

// ---------------- conversion kernels ----------------
__global__ void k_f32_to_bf16(const float* __restrict__ src,
                              unsigned short* __restrict__ dst, int n) {
    int stride = gridDim.x * blockDim.x;
    for (int i = blockIdx.x * blockDim.x + threadIdx.x; i < n; i += stride)
        dst[i] = f2bf(src[i]);
}

// W[k*P+p] (f32, row-major [K,P]) -> Wt[p*K+k] (bf16, [P,K])
__global__ void k_w_transpose_bf16(const float* __restrict__ W,
                                   unsigned short* __restrict__ Wt, int K, int P) {
    int t = blockIdx.x * blockDim.x + threadIdx.x;
    if (t >= K * P) return;
    int k = t / P, p = t - k * P;
    Wt[(size_t)p * K + k] = f2bf(W[t]);
}

// ---------------- WMMA GEMM: Y = Xb[N,256] @ W[256,P] ----------------
// Wt stored transposed [P,256] bf16. One wave per 16x16 tile, K=256 fixed.
__global__ void k_gemm_wmma(const unsigned short* __restrict__ Xb,
                            const unsigned short* __restrict__ Wt0,
                            const unsigned short* __restrict__ Wt1,
                            float* __restrict__ Y0, float* __restrict__ Y1,
                            int N, int P) {
    const int lane  = threadIdx.x & 31;
    const int tileN = blockIdx.x;
    const int tileM = blockIdx.y;
    const unsigned short* Wt = blockIdx.z ? Wt1 : Wt0;
    float*                Y  = blockIdx.z ? Y1  : Y0;
    const int hi = lane >> 4;              // 0: lanes 0-15, 1: lanes 16-31
    int rowA = tileM * 16 + (lane & 15);
    if (rowA >= N) rowA = N - 1;           // clamp; OOB rows never stored
    const int colB = tileN * 16 + (lane & 15);
    const unsigned short* xrow = Xb + (size_t)rowA * 256;
    const unsigned short* wrow = Wt + (size_t)colB * 256;
    v8f c = {};
#pragma unroll
    for (int kk = 0; kk < 256; kk += 32) {
        union { u32x4 u[2]; v16bf v; } a, b;
        // A 16x32 bf16: lanes<16 hold K = kk+{0..7, 16..23}; lanes>=16 shift by 8
        a.u[0] = *(const u32x4*)(xrow + kk + hi * 8);
        a.u[1] = *(const u32x4*)(xrow + kk + hi * 8 + 16);
        // B 32x16 bf16: lanes<16 hold K = kk+{0..7, 16..23}; lanes>=16 shift by 8
        b.u[0] = *(const u32x4*)(wrow + kk + hi * 8);
        b.u[1] = *(const u32x4*)(wrow + kk + hi * 8 + 16);
        c = __builtin_amdgcn_wmma_f32_16x16x32_bf16(false, a.v, false, b.v,
                                                    (short)0, c, false, false);
    }
#pragma unroll
    for (int i = 0; i < 8; ++i) {          // C/D: VGPR i -> M = i + 8*hi
        int m = tileM * 16 + hi * 8 + i;
        if (m < N) Y[(size_t)m * P + colB] = c[i];
    }
}

// ---------------- segment-softmax init ----------------
__global__ void k_seg_init(float* __restrict__ m, float* __restrict__ denom,
                           float* __restrict__ acc, int NH, int NHC) {
    int stride = gridDim.x * blockDim.x;
    int i0 = blockIdx.x * blockDim.x + threadIdx.x;
    for (int j = i0; j < NHC; j += stride) acc[j] = 0.0f;
    for (int j = i0; j < NH;  j += stride) { m[j] = -INFINITY; denom[j] = 0.0f; }
}

// ---------------- edge: logits + segment max ----------------
__global__ void k_edge_logits(const long long* __restrict__ ei,
                              const float* __restrict__ xl, const float* __restrict__ xr,
                              const float* __restrict__ att,
                              float* __restrict__ logits, float* __restrict__ m,
                              int E, int H, int C) {
    int t = blockIdx.x * blockDim.x + threadIdx.x;
    if (t >= E * H) return;
    int e = t / H, h = t - e * H;
    int s = (int)ei[e], d = (int)ei[E + e];
    const int HC = H * C;
    const f32x4* pl = (const f32x4*)(xl + (size_t)s * HC + h * C);
    const f32x4* pr = (const f32x4*)(xr + (size_t)d * HC + h * C);
    const f32x4* pa = (const f32x4*)(att + h * C);
    float acc = 0.0f;
    for (int c4 = 0; c4 < (C >> 2); ++c4) {
        f32x4 l = pl[c4], r = pr[c4], a = pa[c4];
#pragma unroll
        for (int j = 0; j < 4; ++j) {
            float v = l[j] + r[j];
            v = v > 0.0f ? v : 0.2f * v;    // leaky_relu(0.2)
            acc += v * a[j];
        }
    }
    logits[t] = acc;
    atomicMaxF(&m[d * H + h], acc);
}

// ---------------- edge: exp + segment sum ----------------
__global__ void k_edge_exp(const long long* __restrict__ ei,
                           float* __restrict__ logits,
                           const float* __restrict__ m, float* __restrict__ denom,
                           int E, int H) {
    int t = blockIdx.x * blockDim.x + threadIdx.x;
    if (t >= E * H) return;
    int e = t / H, h = t - e * H;
    int d = (int)ei[E + e];
    float mm = m[d * H + h];
    if (mm < -3.0e38f) mm = 0.0f;          // empty-segment guard
    float ex = __expf(logits[t] - mm);
    logits[t] = ex;
    atomicAdd(&denom[d * H + h], ex);
}

// ---------------- edge: weighted scatter ----------------
__global__ void k_edge_scatter(const long long* __restrict__ ei,
                               const float* __restrict__ logits,
                               const float* __restrict__ denom,
                               const float* __restrict__ xl,
                               float* __restrict__ acc,
                               int E, int H, int C) {
    int t = blockIdx.x * blockDim.x + threadIdx.x;
    if (t >= E * H) return;
    int e = t / H, h = t - e * H;
    int s = (int)ei[e], d = (int)ei[E + e];
    const int HC = H * C;
    float alpha = logits[t] / (denom[d * H + h] + 1e-16f);
    const float* pl = xl + (size_t)s * HC + h * C;
    float*       pd = acc + (size_t)d * HC + h * C;
    for (int c = 0; c < C; ++c) atomicAdd(&pd[c], pl[c] * alpha);
}

// ---------------- layer1 finalize: bias + relu + bf16 for next GEMM ----------------
__global__ void k_finalize1(const float* __restrict__ acc, const float* __restrict__ b,
                            unsigned short* __restrict__ hb, int total, int HC) {
    int stride = gridDim.x * blockDim.x;
    for (int i = blockIdx.x * blockDim.x + threadIdx.x; i < total; i += stride) {
        float v = acc[i] + b[i % HC];
        v = v > 0.0f ? v : 0.0f;
        hb[i] = f2bf(v);
    }
}

// ---------------- layer2 finalize: bias + log_softmax (one wave per row) ----------------
__global__ void k_finalize2(float* __restrict__ out, const float* __restrict__ b2, int N) {
    int n = blockIdx.x;
    if (n >= N) return;
    int lane = threadIdx.x;
    float* row = out + (size_t)n * 64;
    float v0 = row[lane]      + b2[lane];
    float v1 = row[lane + 32] + b2[lane + 32];
    row[lane]      = v0;
    row[lane + 32] = v1;
    float mx = fmaxf(v0, v1);
#pragma unroll
    for (int o = 16; o > 0; o >>= 1) mx = fmaxf(mx, __shfl_xor(mx, o, 32));
    float s = __expf(v0 - mx) + __expf(v1 - mx);
#pragma unroll
    for (int o = 16; o > 0; o >>= 1) s += __shfl_xor(s, o, 32);
    float lse = mx + __logf(s);
    float* row2 = out + (size_t)N * 64 + (size_t)n * 64;
    row2[lane]      = v0 - lse;
    row2[lane + 32] = v1 - lse;
}

extern "C" void kernel_launch(void* const* d_in, const int* in_sizes, int n_in,
                              void* d_out, int out_size, void* d_ws, size_t ws_size,
                              hipStream_t stream) {
    (void)n_in; (void)out_size; (void)ws_size;
    const float*     x    = (const float*)d_in[0];
    const long long* ei   = (const long long*)d_in[1];
    const float*     Wl1  = (const float*)d_in[2];
    const float*     Wr1  = (const float*)d_in[3];
    const float*     att1 = (const float*)d_in[4];
    const float*     b1   = (const float*)d_in[5];
    const float*     Wl2  = (const float*)d_in[6];
    const float*     Wr2  = (const float*)d_in[7];
    const float*     att2 = (const float*)d_in[8];
    const float*     b2   = (const float*)d_in[9];
    float* out = (float*)d_out;

    const int F = 256, H1 = 8, C1 = 32, HC = 256, OUT = 64;
    const int N = in_sizes[0] / F;
    const int E = in_sizes[1] / 2;

    char* ws = (char*)d_ws;
    size_t off = 0;
    auto take = [&](size_t bytes) -> char* {
        char* p = ws + off;
        off = (off + bytes + 255) & ~(size_t)255;
        return p;
    };
    unsigned short* Xb     = (unsigned short*)take((size_t)N * F * 2);   // bf16 layer input
    unsigned short* Wlt    = (unsigned short*)take((size_t)HC * F * 2);  // bf16 Wt, max 256x256
    unsigned short* Wrt    = (unsigned short*)take((size_t)HC * F * 2);
    float*          XL     = (float*)take((size_t)N * HC * 4);
    float*          XR     = (float*)take((size_t)N * HC * 4);
    float*          logits = (float*)take((size_t)E * H1 * 4);
    float*          mbuf   = (float*)take((size_t)N * H1 * 4);
    float*          dbuf   = (float*)take((size_t)N * H1 * 4);
    float*          acc1   = (float*)take((size_t)N * HC * 4);

    // ================= layer 1 =================
    k_f32_to_bf16<<<4096, 256, 0, stream>>>(x, Xb, N * F);
    k_w_transpose_bf16<<<(F * HC + 255) / 256, 256, 0, stream>>>(Wl1, Wlt, F, HC);
    k_w_transpose_bf16<<<(F * HC + 255) / 256, 256, 0, stream>>>(Wr1, Wrt, F, HC);
    {
        dim3 g(HC / 16, (N + 15) / 16, 2);
        k_gemm_wmma<<<g, 32, 0, stream>>>(Xb, Wlt, Wrt, XL, XR, N, HC);
    }
    k_seg_init<<<4096, 256, 0, stream>>>(mbuf, dbuf, acc1, N * H1, N * HC);
    {
        int T = E * H1, blk = (T + 255) / 256;
        k_edge_logits <<<blk, 256, 0, stream>>>(ei, XL, XR, att1, logits, mbuf, E, H1, C1);
        k_edge_exp    <<<blk, 256, 0, stream>>>(ei, logits, mbuf, dbuf, E, H1);
        k_edge_scatter<<<blk, 256, 0, stream>>>(ei, logits, dbuf, XL, acc1, E, H1, C1);
    }
    k_finalize1<<<4096, 256, 0, stream>>>(acc1, b1, Xb, N * HC, HC);

    // ================= layer 2 =================
    k_w_transpose_bf16<<<(HC * OUT + 255) / 256, 256, 0, stream>>>(Wl2, Wlt, HC, OUT);
    k_w_transpose_bf16<<<(HC * OUT + 255) / 256, 256, 0, stream>>>(Wr2, Wrt, HC, OUT);
    {
        dim3 g(OUT / 16, (N + 15) / 16, 2);
        k_gemm_wmma<<<g, 32, 0, stream>>>(Xb, Wlt, Wrt, XL, XR, N, OUT);
    }
    k_seg_init<<<4096, 256, 0, stream>>>(mbuf, dbuf, out, N, N * OUT);
    {
        int T = E, blk = (T + 255) / 256;
        k_edge_logits <<<blk, 256, 0, stream>>>(ei, XL, XR, att2, logits, mbuf, E, 1, OUT);
        k_edge_exp    <<<blk, 256, 0, stream>>>(ei, logits, mbuf, dbuf, E, 1);
        k_edge_scatter<<<blk, 256, 0, stream>>>(ei, logits, dbuf, XL, out, E, 1, OUT);
    }
    k_finalize2<<<N, 32, 0, stream>>>(out, b2, N);
}
